// MultiHeadAttention_36644660969649
// MI455X (gfx1250) — compile-verified, matrix operand order checked by reading, not run on hardware
//
#include <hip/hip_runtime.h>
#include <hip/hip_bf16.h>
#include <math.h>

typedef __bf16 bf16;
typedef __attribute__((ext_vector_type(16))) __bf16 bf16x16;
typedef __attribute__((ext_vector_type(8)))  float  f32x8;
typedef __attribute__((__vector_size__(4 * sizeof(int)))) int v4i;

#define AS1 __attribute__((address_space(1)))
#define AS3 __attribute__((address_space(3)))

#if defined(__gfx1250__) && __has_builtin(__builtin_amdgcn_global_load_async_to_lds_b128)
#define HAVE_ASYNC 1
#endif

// Async (or fallback sync) 16-byte global -> LDS copy.
__device__ __forceinline__ void cp16_g2l(const bf16* g, bf16* l) {
#if defined(HAVE_ASYNC)
  __builtin_amdgcn_global_load_async_to_lds_b128(
      (AS1 v4i*)const_cast<bf16*>(g), (AS3 v4i*)l, 0, 0);
#else
  *(uint4*)l = *(const uint4*)g;
#endif
}

// Wait until at most N async copies are outstanding (pipelined staging).
template <int N>
__device__ __forceinline__ void cp_wait() {
#if defined(HAVE_ASYNC)
#if __has_builtin(__builtin_amdgcn_s_wait_asynccnt)
  __builtin_amdgcn_s_wait_asynccnt(N);
#else
  asm volatile("s_wait_asynccnt %0" :: "i"(N) : "memory");
#endif
#endif
}

// Load 16 contiguous bf16 (32B) from a 16B-aligned address as two uint4s.
__device__ __forceinline__ bf16x16 ld16(const bf16* p) {
  const uint4* q = (const uint4*)p;
  union { uint4 u[2]; bf16x16 v; } tmp;
  tmp.u[0] = q[0];
  tmp.u[1] = q[1];
  return tmp.v;
}

// ---------------------------------------------------------------------------
// fp32 -> bf16 elementwise conversion
// ---------------------------------------------------------------------------
__global__ void cvt_f32_bf16(const float* __restrict__ in, bf16* __restrict__ out, int n) {
  int i = blockIdx.x * blockDim.x + threadIdx.x;
  int stride = gridDim.x * blockDim.x;
  for (; i < n; i += stride) out[i] = (bf16)in[i];
}

// ---------------------------------------------------------------------------
// fp32 [K][N] -> bf16 [N][K] transposed conversion (LDS-tiled 64x64).
// ---------------------------------------------------------------------------
__global__ __launch_bounds__(256) void cvt_transpose_bf16(
    const float* __restrict__ in, bf16* __restrict__ out, int K, int N)
{
  __shared__ bf16 Ts[64][72];
  const int n0 = blockIdx.x * 64, k0 = blockIdx.y * 64;
  const int t = threadIdx.x;

  #pragma unroll
  for (int i = 0; i < 4; ++i) {
    int c = i * 256 + t;              // 1024 float4 chunks
    int r = c >> 4, ch = c & 15;
    float4 v = *(const float4*)(in + (size_t)(k0 + r) * N + n0 + ch * 4);
    Ts[r][ch * 4 + 0] = (bf16)v.x;
    Ts[r][ch * 4 + 1] = (bf16)v.y;
    Ts[r][ch * 4 + 2] = (bf16)v.z;
    Ts[r][ch * 4 + 3] = (bf16)v.w;
  }
  __syncthreads();

  #pragma unroll
  for (int i = 0; i < 2; ++i) {
    int c = i * 256 + t;              // 512 8-bf16 chunks
    int r = c >> 3, ch = c & 7;       // r = n offset, ch*8 = k offset
    union { bf16 h[8]; uint4 u; } tmp;
    #pragma unroll
    for (int j = 0; j < 8; ++j) tmp.h[j] = Ts[ch * 8 + j][r];
    *(uint4*)(out + (size_t)(n0 + r) * K + k0 + ch * 8) = tmp.u;
  }
}

// ---------------------------------------------------------------------------
// Tiled bf16 WMMA GEMM: C[M,N] = A[M,K] @ Bt[N,K]^T + bias.
// Tile 128x128, BK=32, 256 threads = 8 waves, each wave a 32x64 sub-tile
// (8 WMMAs per K-step). Double-buffered async LDS staging: tile k+1's DMA
// runs under tile k's WMMAs (wait asynccnt<=4 drains only the older group).
// ---------------------------------------------------------------------------
#define BM 128
#define BN 128
#define BK 32
#define LDT 40                 // 32 + 8 bf16 pad (80B rows, 16B aligned)
#define TSZ (BM * LDT)         // one LDS tile in bf16 elements

__global__ __launch_bounds__(256) void gemm_bf16_wmma(
    const bf16* __restrict__ A,      // [M][K]
    const bf16* __restrict__ Bt,     // [N][K] (pre-transposed weights)
    const float* __restrict__ bias,
    float* Cf, bf16* Cb,
    int M, int N, int K, int applyRope, int storeTransposed)
{
  __shared__ bf16 As[2 * TSZ];
  __shared__ bf16 Bs[2 * TSZ];

  const int t = threadIdx.x, lane = t & 31, w = t >> 5;
  const int wr = w & 3, wc = w >> 2;       // wave row (32 rows), wave col (64 cols)
  const int half = lane >> 4, lr = lane & 15;
  const int m0 = blockIdx.y * BM, n0 = blockIdx.x * BN;

  // Per-thread staging chunks: 512 16B chunks per 128x32 tile, 2 per thread.
  const int c0 = t, c1 = t + 256;
  const int ra0 = c0 >> 2, ka0 = (c0 & 3) * 8;
  const int ra1 = c1 >> 2, ka1 = (c1 & 3) * 8;
  const int oA0 = ra0 * LDT + ka0, oA1 = ra1 * LDT + ka1;
  const bf16* gA0 = A  + (size_t)(m0 + ra0) * K + ka0;
  const bf16* gA1 = A  + (size_t)(m0 + ra1) * K + ka1;
  const bf16* gB0 = Bt + (size_t)(n0 + ra0) * K + ka0;
  const bf16* gB1 = Bt + (size_t)(n0 + ra1) * K + ka1;

  f32x8 acc[2][4] = {};
  const int nk = K / BK;

  // Prologue: stage tile 0 into buffer 0.
  cp16_g2l(gA0, As + oA0);
  cp16_g2l(gA1, As + oA1);
  cp16_g2l(gB0, Bs + oA0);
  cp16_g2l(gB1, Bs + oA1);
  gA0 += BK; gA1 += BK; gB0 += BK; gB1 += BK;

  for (int kt = 0; kt < nk; ++kt) {
    const int cob = (kt & 1) * TSZ;       // compute buffer offset
    const int stb = TSZ - cob;            // staging buffer offset
    if (kt + 1 < nk) {
      cp16_g2l(gA0, As + stb + oA0);
      cp16_g2l(gA1, As + stb + oA1);
      cp16_g2l(gB0, Bs + stb + oA0);
      cp16_g2l(gB1, Bs + stb + oA1);
      if (kt + 2 < nk) {
        __builtin_prefetch(gA0 + BK, 0, 3);
        __builtin_prefetch(gB0 + BK, 0, 3);
      }
      gA0 += BK; gA1 += BK; gB0 += BK; gB1 += BK;
      cp_wait<4>();                       // drain tile kt only; kt+1 in flight
    } else {
      cp_wait<0>();
    }
    __syncthreads();

    bf16x16 a0 = ld16(&As[cob + (wr * 32 +      lr) * LDT + half * 16]);
    bf16x16 a1 = ld16(&As[cob + (wr * 32 + 16 + lr) * LDT + half * 16]);
    #pragma unroll
    for (int j = 0; j < 4; ++j) {
      bf16x16 b = ld16(&Bs[cob + (wc * 64 + j * 16 + lr) * LDT + half * 16]);
      acc[0][j] = __builtin_amdgcn_wmma_f32_16x16x32_bf16(
          false, a0, false, b, (short)0, acc[0][j], false, false);
      acc[1][j] = __builtin_amdgcn_wmma_f32_16x16x32_bf16(
          false, a1, false, b, (short)0, acc[1][j], false, false);
    }
    __syncthreads();                      // protect buffer before next overwrite
  }

  // Epilogue: bias (+ optional RoPE), store bf16 / f32, optionally transposed.
  const float RLOG = 8.517193191416238f / 64.f;  // ln(5000)/gvs
  #pragma unroll
  for (int g = 0; g < 2; ++g) {
    #pragma unroll
    for (int j = 0; j < 4; ++j) {
      #pragma unroll
      for (int r = 0; r < 8; ++r) {
        int row = m0 + wr * 32 + g * 16 + r + 8 * half;  // C-layout M
        int col = n0 + wc * 64 + j * 16 + lr;            // C-layout N
        float v = acc[g][j][r] + bias[col];
        if (applyRope) {
          int p = (col & 127) >> 1;                       // pair idx in head dim
          float phi = __expf(-(float)p * RLOG);
          float s, c;
          __sincosf((float)row * phi, &s, &c);
          float partner = __shfl_xor(v, 1, 32);           // adjacent col = adjacent lane
          v = (col & 1) ? (partner * s + v * c) : (v * c - partner * s);
        }
        if (Cb) {
          if (storeTransposed) Cb[(size_t)col * M + row] = (bf16)v;
          else                 Cb[(size_t)row * N + col] = (bf16)v;
        } else {
          Cf[(size_t)row * N + col] = v;
        }
      }
    }
  }
}

// ---------------------------------------------------------------------------
// Causal flash attention, bf16 WMMA, GQA 4:1. One (head, 64-query block) per
// workgroup, 4 waves. Double-buffered async K/V staging (16 copies per wave
// per tile -> wait asynccnt<=16 keeps the next tile's DMA in flight).
// ---------------------------------------------------------------------------
#define LKS 136                 // K tile row stride (128 + 8)
#define LVS 72                  // V^T row stride (64 + 8)
#define LPS 72                  // P scratch row stride
#define KSZ (64 * LKS)
#define VSZ (128 * LVS)

__global__ __launch_bounds__(128) void attn_fwd(
    const bf16* __restrict__ Qb,   // [S][4096]
    const bf16* __restrict__ Kb,   // [S][1024]
    const bf16* __restrict__ Vt,   // [1024][S] (transposed)
    bf16* __restrict__ Ob)         // [S][4096]
{
  __shared__ bf16 Ks[2 * KSZ];
  __shared__ bf16 Vs[2 * VSZ];
  __shared__ bf16 Ps[4 * 16 * LPS];

  const int h = blockIdx.x, qb = blockIdx.y;
  const int t = threadIdx.x, lane = t & 31, w = t >> 5;
  const int half = lane >> 4, lr = lane & 15;
  const int kh = h >> 2;
  const int qbase = qb * 64;
  const int r0 = qbase + w * 16;
  const int S = 2048;

  bf16x16 qf[4];
  #pragma unroll
  for (int kk = 0; kk < 4; ++kk)
    qf[kk] = ld16(Qb + (size_t)(r0 + lr) * 4096 + h * 128 + kk * 32 + half * 16);

  f32x8 o[8] = {};
  float m[8], l[8];
  #pragma unroll
  for (int r = 0; r < 8; ++r) { m[r] = -INFINITY; l[r] = 0.f; }

  bf16* Pw = Ps + w * 16 * LPS;
  const float sc = 0.08838834764831843f;  // 1/sqrt(128)

  // Stage one K/V tile (async, 16 copies per wave).
  auto stageKV = [&](int kbase, int buf) {
    bf16* kd = Ks + buf * KSZ;
    bf16* vd = Vs + buf * VSZ;
    #pragma unroll
    for (int i = 0; i < 8; ++i) {
      int c = i * 128 + t;
      int row = c >> 4, ch = c & 15;
      cp16_g2l(Kb + (size_t)(kbase + row) * 1024 + kh * 128 + ch * 8,
               kd + row * LKS + ch * 8);
    }
    #pragma unroll
    for (int i = 0; i < 8; ++i) {
      int c = i * 128 + t;
      int row = c >> 3, ch = c & 7;   // row = d, ch*8 = kcol offset
      cp16_g2l(Vt + (size_t)(kh * 128 + row) * S + kbase + ch * 8,
               vd + row * LVS + ch * 8);
    }
  };

  stageKV(0, 0);  // prologue

  for (int kb = 0; kb <= qb; ++kb) {
    const int cur = kb & 1;
    const int kbase = kb * 64;
    if (kb + 1 <= qb) {
      stageKV((kb + 1) * 64, cur ^ 1);
      cp_wait<16>();                      // drain tile kb only
    } else {
      cp_wait<0>();
    }
    __syncthreads();

    const bf16* kk_base = Ks + cur * KSZ;
    const bf16* vv_base = Vs + cur * VSZ;

    // S = Q @ K^T
    f32x8 sf[4] = {};
    #pragma unroll
    for (int nt = 0; nt < 4; ++nt) {
      #pragma unroll
      for (int kk = 0; kk < 4; ++kk) {
        bf16x16 b = ld16(&kk_base[(nt * 16 + lr) * LKS + kk * 32 + half * 16]);
        sf[nt] = __builtin_amdgcn_wmma_f32_16x16x32_bf16(
            false, qf[kk], false, b, (short)0, sf[nt], false, false);
      }
    }

    // Scale + causal mask + row max.
    float rm[8];
    #pragma unroll
    for (int r = 0; r < 8; ++r) rm[r] = -INFINITY;
    #pragma unroll
    for (int nt = 0; nt < 4; ++nt) {
      #pragma unroll
      for (int r = 0; r < 8; ++r) {
        int qrow = r0 + r + 8 * half;
        int kcol = kbase + nt * 16 + lr;
        float v = sf[nt][r] * sc;
        v = (kcol <= qrow) ? v : -INFINITY;
        sf[nt][r] = v;
        rm[r] = fmaxf(rm[r], v);
      }
    }
    #pragma unroll
    for (int r = 0; r < 8; ++r) {
      #pragma unroll
      for (int d = 1; d < 16; d <<= 1)
        rm[r] = fmaxf(rm[r], __shfl_xor(rm[r], d, 32));
    }

    // Online softmax rescale.
    float rs[8];
    #pragma unroll
    for (int r = 0; r < 8; ++r) {
      float mn = fmaxf(m[r], rm[r]);
      float corr = __expf(m[r] - mn);
      m[r] = mn;
      l[r] *= corr;
      #pragma unroll
      for (int nt2 = 0; nt2 < 8; ++nt2) o[nt2][r] *= corr;
      rs[r] = 0.f;
    }

    // P = exp(S - m) -> per-wave LDS (C-frag -> A-frag relayout).
    #pragma unroll
    for (int nt = 0; nt < 4; ++nt) {
      #pragma unroll
      for (int r = 0; r < 8; ++r) {
        float p = __expf(sf[nt][r] - m[r]);
        rs[r] += p;
        Pw[(r + 8 * half) * LPS + nt * 16 + lr] = (bf16)p;
      }
    }
    #pragma unroll
    for (int r = 0; r < 8; ++r) {
      #pragma unroll
      for (int d = 1; d < 16; d <<= 1)
        rs[r] += __shfl_xor(rs[r], d, 32);
      l[r] += rs[r];
    }

    // O += P @ V
    #pragma unroll
    for (int ks = 0; ks < 2; ++ks) {
      bf16x16 a = ld16(&Pw[lr * LPS + ks * 32 + half * 16]);
      #pragma unroll
      for (int nt2 = 0; nt2 < 8; ++nt2) {
        bf16x16 b = ld16(&vv_base[(nt2 * 16 + lr) * LVS + ks * 32 + half * 16]);
        o[nt2] = __builtin_amdgcn_wmma_f32_16x16x32_bf16(
            false, a, false, b, (short)0, o[nt2], false, false);
      }
    }
    __syncthreads();                      // protect buffer before next overwrite
  }

  // Normalize and store (bf16, feeds the Wo GEMM).
  #pragma unroll
  for (int nt2 = 0; nt2 < 8; ++nt2) {
    #pragma unroll
    for (int r = 0; r < 8; ++r) {
      int row = r0 + r + 8 * half;
      int col = h * 128 + nt2 * 16 + lr;
      Ob[(size_t)row * 4096 + col] = (bf16)(o[nt2][r] / l[r]);
    }
  }
}

// ---------------------------------------------------------------------------
// Launch
// ---------------------------------------------------------------------------
extern "C" void kernel_launch(void* const* d_in, const int* in_sizes, int n_in,
                              void* d_out, int out_size, void* d_ws, size_t ws_size,
                              hipStream_t stream) {
  (void)in_sizes; (void)n_in; (void)out_size; (void)ws_size;

  const int S = 2048, D = 4096, GS = 1024, NH = 32;

  const float* x  = (const float*)d_in[0];
  const float* y  = (const float*)d_in[1];
  const float* Wq = (const float*)d_in[2];
  const float* bq = (const float*)d_in[3];
  const float* Wk = (const float*)d_in[4];
  const float* bk = (const float*)d_in[5];
  const float* Wv = (const float*)d_in[6];
  const float* bv = (const float*)d_in[7];
  const float* Wo = (const float*)d_in[8];
  const float* bo = (const float*)d_in[9];

  char* ws = (char*)d_ws;
  size_t off = 0;
  auto alloc = [&](size_t bytes) -> char* {
    char* p = ws + off;
    off = (off + bytes + 255) & ~(size_t)255;
    return p;
  };
  bf16* xb   = (bf16*)alloc((size_t)S * D * 2);
  bf16* yb   = (bf16*)alloc((size_t)S * D * 2);
  bf16* Wqt  = (bf16*)alloc((size_t)D * D * 2);   // [D][D]   (N,K)
  bf16* Wkt  = (bf16*)alloc((size_t)GS * D * 2);  // [GS][D]
  bf16* Wvt  = (bf16*)alloc((size_t)GS * D * 2);
  bf16* Wot  = (bf16*)alloc((size_t)D * D * 2);
  bf16* Qbf  = (bf16*)alloc((size_t)S * D * 2);
  bf16* Kbf  = (bf16*)alloc((size_t)S * GS * 2);
  bf16* Vtb  = (bf16*)alloc((size_t)GS * S * 2);  // V transposed [GS][S]
  bf16* Abf  = (bf16*)alloc((size_t)S * D * 2);

  // 1) fp32 -> bf16 (activations elementwise, weights transposed-tiled)
  cvt_f32_bf16<<<2048, 256, 0, stream>>>(x, xb, S * D);
  cvt_f32_bf16<<<2048, 256, 0, stream>>>(y, yb, S * D);
  cvt_transpose_bf16<<<dim3(D / 64,  D / 64), 256, 0, stream>>>(Wq, Wqt, D, D);
  cvt_transpose_bf16<<<dim3(GS / 64, D / 64), 256, 0, stream>>>(Wk, Wkt, D, GS);
  cvt_transpose_bf16<<<dim3(GS / 64, D / 64), 256, 0, stream>>>(Wv, Wvt, D, GS);
  cvt_transpose_bf16<<<dim3(D / 64,  D / 64), 256, 0, stream>>>(Wo, Wot, D, D);

  // 2) QKV projections (+bias; RoPE fused on Q,K; V stored transposed)
  gemm_bf16_wmma<<<dim3(D / BN, S / BM), 256, 0, stream>>>(
      xb, Wqt, bq, nullptr, Qbf, S, D, D, 1, 0);
  gemm_bf16_wmma<<<dim3(GS / BN, S / BM), 256, 0, stream>>>(
      yb, Wkt, bk, nullptr, Kbf, S, GS, D, 1, 0);
  gemm_bf16_wmma<<<dim3(GS / BN, S / BM), 256, 0, stream>>>(
      yb, Wvt, bv, nullptr, Vtb, S, GS, D, 0, 1);

  // 3) Causal flash attention
  attn_fwd<<<dim3(NH, S / 64), 128, 0, stream>>>(Qbf, Kbf, Vtb, Abf);

  // 4) Output projection -> f32 result
  gemm_bf16_wmma<<<dim3(D / BN, S / BM), 256, 0, stream>>>(
      Abf, Wot, bo, (float*)d_out, nullptr, S, D, D, 0, 0);
}
